// Decoder_64699387347471
// MI455X (gfx1250) — matrix operand order repaired and checked by reading, not compiled, not run on hardware
//
#include <hip/hip_runtime.h>
#include <hip/hip_bf16.h>
#include <math.h>

// ---------------------------------------------------------------------------
// MI455X attention-LSTM decoder, WMMA bf16 (f32 accum).
//  - Fragments loaded as contiguous b128 spans per the ISA VGPR layouts.
//  - One wave owns a 16x64 D tile: 1 A-frag + 4 B-frags + 4 WMMAs per K step.
//  - K loops software-pipelined (next-step fragments loaded before current
//    WMMAs) so b128 loads overlap the matrix pipe instead of s_wait 0.
//  - attn_img kept bf16 (81 MB) => L2-resident (192 MB).
// ---------------------------------------------------------------------------

typedef __attribute__((ext_vector_type(16))) __bf16 v16bf;
typedef __attribute__((ext_vector_type(8)))  float  v8f;

#define B_   256
#define P_   620
#define C_   512
#define T_   32
#define V_   140
#define H_   256
#define A_   256
#define G_   1024   // 4*H
#define KX_  160    // V padded to 32-multiple (zero filled)
#define NCD_ 144    // V padded to 16-multiple for out GEMM tiles

__device__ __forceinline__ __bf16 f2bf(float f) {
  unsigned u = __builtin_bit_cast(unsigned, f);
  unsigned r = (u + 0x7FFFu + ((u >> 16) & 1u)) >> 16;   // round-to-nearest-even
  return __builtin_bit_cast(__bf16, (unsigned short)r);
}
__device__ __forceinline__ float bf2f(__bf16 b) {
  unsigned u = ((unsigned)__builtin_bit_cast(unsigned short, b)) << 16;
  return __builtin_bit_cast(float, u);
}
__device__ __forceinline__ float bflo(unsigned w) { return __builtin_bit_cast(float, w << 16); }
__device__ __forceinline__ float bfhi(unsigned w) { return __builtin_bit_cast(float, w & 0xFFFF0000u); }

union FragU { uint4 q[2]; v16bf v; };

// A fragment: two contiguous 16B spans (ISA 7.12.2 16-bit A 16x32 layout:
// lane<16 holds K[0..7]+K[16..23]; lane>=16 holds K[8..15]+K[24..31]).
__device__ __forceinline__ v16bf load_afrag(const __bf16* __restrict__ Arow, int kb, int half) {
  FragU u;
  u.q[0] = *(const uint4*)(Arow + kb + 8 * half);
  u.q[1] = *(const uint4*)(Arow + kb + 16 + 8 * half);
  return u.v;
}
// B fragment: two contiguous 16B spans (K = 8*half .. +8, 16 + 8*half .. +8).
__device__ __forceinline__ v16bf load_bfrag(const __bf16* __restrict__ Wrow, int kb, int half) {
  FragU u;
  u.q[0] = *(const uint4*)(Wrow + kb + 8 * half);
  u.q[1] = *(const uint4*)(Wrow + kb + 16 + 8 * half);
  return u.v;
}
// A fragment from fp32 source, converted on the fly (vector loads; the
// conversion VALU co-executes with the XDL WMMA pipe).
__device__ __forceinline__ v16bf load_afrag_f32(const float* __restrict__ Arow, int kb, int half) {
  const float4* p0 = (const float4*)(Arow + kb + 8 * half);
  const float4* p1 = (const float4*)(Arow + kb + 16 + 8 * half);
  float4 a = p0[0], b = p0[1], c = p1[0], d = p1[1];
  v16bf v;
  v[0] = f2bf(a.x);  v[1] = f2bf(a.y);  v[2] = f2bf(a.z);  v[3] = f2bf(a.w);
  v[4] = f2bf(b.x);  v[5] = f2bf(b.y);  v[6] = f2bf(b.z);  v[7] = f2bf(b.w);
  v[8] = f2bf(c.x);  v[9] = f2bf(c.y);  v[10] = f2bf(c.z); v[11] = f2bf(c.w);
  v[12] = f2bf(d.x); v[13] = f2bf(d.y); v[14] = f2bf(d.z); v[15] = f2bf(d.w);
  return v;
}

__device__ __forceinline__ v8f wmma_bf16(v16bf a, v16bf b, v8f c) {
  return __builtin_amdgcn_wmma_f32_16x16x32_bf16(false, a, false, b, (short)0, c, false, false);
}

// 16x64 tile over one K segment, double-buffered: next-step fragments are
// issued before the current WMMAs so loads stay in flight under the MMA pipe.
__device__ __forceinline__ void gemm_seg16x64(const __bf16* __restrict__ Arow,
                                              const __bf16* __restrict__ w0,
                                              const __bf16* __restrict__ w1,
                                              const __bf16* __restrict__ w2,
                                              const __bf16* __restrict__ w3,
                                              int K, int half, v8f acc[4]) {
  v16bf af = load_afrag(Arow, 0, half);
  v16bf b0 = load_bfrag(w0, 0, half);
  v16bf b1 = load_bfrag(w1, 0, half);
  v16bf b2 = load_bfrag(w2, 0, half);
  v16bf b3 = load_bfrag(w3, 0, half);
  for (int kb = 0; kb < K; kb += 32) {
    int kn = (kb + 32 < K) ? kb + 32 : 0;        // branchless wrap (in-bounds)
    v16bf afn = load_afrag(Arow, kn, half);
    v16bf b0n = load_bfrag(w0, kn, half);
    v16bf b1n = load_bfrag(w1, kn, half);
    v16bf b2n = load_bfrag(w2, kn, half);
    v16bf b3n = load_bfrag(w3, kn, half);
    acc[0] = wmma_bf16(af, b0, acc[0]);
    acc[1] = wmma_bf16(af, b1, acc[1]);
    acc[2] = wmma_bf16(af, b2, acc[2]);
    acc[3] = wmma_bf16(af, b3, acc[3]);
    af = afn; b0 = b0n; b1 = b1n; b2 = b2n; b3 = b3n;
  }
}

// ---------------- setup kernels ----------------

__global__ void k_cvt(const float* __restrict__ src, __bf16* __restrict__ dst,
                      int N, int K, int srcStride, int srcOff, int Npad, int Kpad) {
  int idx = blockIdx.x * blockDim.x + threadIdx.x;
  if (idx >= Npad * Kpad) return;
  int n = idx / Kpad, k = idx % Kpad;
  float v = (n < N && k < K) ? src[(size_t)n * srcStride + srcOff + k] : 0.0f;
  dst[idx] = f2bf(v);
}

__global__ void k_init_state(const float* __restrict__ targets,
                             float* __restrict__ h, float* __restrict__ c,
                             __bf16* __restrict__ hbf, __bf16* __restrict__ xbf) {
  int idx = blockIdx.x * blockDim.x + threadIdx.x;   // covers B*H
  if (idx < B_ * H_) { h[idx] = 0.f; c[idx] = 0.f; hbf[idx] = f2bf(0.f); }
  if (idx < B_ * KX_) {
    int b = idx / KX_, k = idx % KX_;
    float v = (k < V_) ? targets[(size_t)b * T_ * V_ + k] : 0.0f;  // x0 = targets[:,0,:]
    xbf[idx] = f2bf(v);
  }
}

// ---------------- GEMM kernels (WMMA) ----------------

// attn[b,p,a] = img[b,p,:] . Ic_w[a,:] + Ic_b[a], bf16 out.  M=B*P, K=512, N=256.
__global__ void k_attn_gemm(const float* __restrict__ img, const __bf16* __restrict__ wIc,
                            const float* __restrict__ Ic_b, __bf16* __restrict__ attn) {
  const int lane = threadIdx.x & 31, wave = threadIdx.x >> 5;
  const int half = lane >> 4, l = lane & 15;
  int tile = blockIdx.x * 8 + wave;            // (B*P/16)*(A/64) tiles
  int ng = tile & 3, mt = tile >> 2;           // 4 N-groups of 64
  int m0 = mt * 16, n0 = ng * 64;

  const float* Ap = img + (size_t)(m0 + l) * C_;
  const __bf16* w0 = wIc + (size_t)(n0 + l) * C_;
  const __bf16* w1 = w0 + (size_t)16 * C_;
  const __bf16* w2 = w0 + (size_t)32 * C_;
  const __bf16* w3 = w0 + (size_t)48 * C_;
  v8f acc[4] = {};
  v16bf af = load_afrag_f32(Ap, 0, half);
  v16bf b0 = load_bfrag(w0, 0, half);
  v16bf b1 = load_bfrag(w1, 0, half);
  v16bf b2 = load_bfrag(w2, 0, half);
  v16bf b3 = load_bfrag(w3, 0, half);
  for (int kb = 0; kb < C_; kb += 32) {
    int kn = (kb + 32 < C_) ? kb + 32 : 0;
    v16bf afn = load_afrag_f32(Ap, kn, half);
    v16bf b0n = load_bfrag(w0, kn, half);
    v16bf b1n = load_bfrag(w1, kn, half);
    v16bf b2n = load_bfrag(w2, kn, half);
    v16bf b3n = load_bfrag(w3, kn, half);
    acc[0] = wmma_bf16(af, b0, acc[0]);
    acc[1] = wmma_bf16(af, b1, acc[1]);
    acc[2] = wmma_bf16(af, b2, acc[2]);
    acc[3] = wmma_bf16(af, b3, acc[3]);
    af = afn; b0 = b0n; b1 = b1n; b2 = b2n; b3 = b3n;
  }
#pragma unroll
  for (int i = 0; i < 4; ++i) {
    int n = n0 + 16 * i + l;
    float bias = Ic_b[n];
#pragma unroll
    for (int r = 0; r < 8; ++r) {
      int m = m0 + r + 8 * half;               // C/D layout: row = r + 8*(lane>=16)
      attn[(size_t)m * A_ + n] = f2bf(acc[i][r] + bias);
    }
  }
}

// q = h @ Hc_w^T + Hc_b.  M=B=256, K=H=256, N=A=256. 16x64 tiles.
__global__ void k_q_gemm(const __bf16* __restrict__ hbf, const __bf16* __restrict__ wHc,
                         const float* __restrict__ Hc_b, float* __restrict__ q) {
  const int lane = threadIdx.x & 31, wave = threadIdx.x >> 5;
  const int half = lane >> 4, l = lane & 15;
  int tile = blockIdx.x * 8 + wave;            // 16 mt * 4 ngroups = 64 waves
  int ng = tile & 3, mt = tile >> 2;
  int m0 = mt * 16, n0 = ng * 64;
  const __bf16* Ar = hbf + (size_t)(m0 + l) * H_;
  const __bf16* w0 = wHc + (size_t)(n0 + l) * H_;
  v8f acc[4] = {};
  gemm_seg16x64(Ar, w0, w0 + 16 * H_, w0 + 32 * H_, w0 + 48 * H_, H_, half, acc);
#pragma unroll
  for (int i = 0; i < 4; ++i) {
    int n = n0 + 16 * i + l;
    float bias = Hc_b[n];
#pragma unroll
    for (int r = 0; r < 8; ++r)
      q[(size_t)(m0 + r + 8 * half) * A_ + n] = acc[i][r] + bias;
  }
}

// gates = [x|ctx] @ W_ih^T + h @ W_hh^T + biases.  M=256, N=1024, K=160+256+256.
__global__ void k_gates_gemm(const __bf16* __restrict__ xbf, const __bf16* __restrict__ ctxbf,
                             const __bf16* __restrict__ hbf,
                             const __bf16* __restrict__ wWx, const __bf16* __restrict__ wWc,
                             const __bf16* __restrict__ wWhh,
                             const float* __restrict__ b_ih, const float* __restrict__ b_hh,
                             float* __restrict__ gates) {
  const int lane = threadIdx.x & 31, wave = threadIdx.x >> 5;
  const int half = lane >> 4, l = lane & 15;
  int tile = blockIdx.x * 8 + wave;            // 16 mt * 16 ngroups = 256 waves
  int ng = tile & 15, mt = tile >> 4;
  int m0 = mt * 16, n0 = ng * 64;
  v8f acc[4] = {};
  {
    const __bf16* Ar = xbf + (size_t)(m0 + l) * KX_;
    const __bf16* w0 = wWx + (size_t)(n0 + l) * KX_;
    gemm_seg16x64(Ar, w0, w0 + 16 * KX_, w0 + 32 * KX_, w0 + 48 * KX_, KX_, half, acc);
  }
  {
    const __bf16* Ar = ctxbf + (size_t)(m0 + l) * A_;
    const __bf16* w0 = wWc + (size_t)(n0 + l) * A_;
    gemm_seg16x64(Ar, w0, w0 + 16 * A_, w0 + 32 * A_, w0 + 48 * A_, A_, half, acc);
  }
  {
    const __bf16* Ar = hbf + (size_t)(m0 + l) * H_;
    const __bf16* w0 = wWhh + (size_t)(n0 + l) * H_;
    gemm_seg16x64(Ar, w0, w0 + 16 * H_, w0 + 32 * H_, w0 + 48 * H_, H_, half, acc);
  }
#pragma unroll
  for (int i = 0; i < 4; ++i) {
    int n = n0 + 16 * i + l;
    float bias = b_ih[n] + b_hh[n];
#pragma unroll
    for (int r = 0; r < 8; ++r)
      gates[(size_t)(m0 + r + 8 * half) * G_ + n] = acc[i][r] + bias;
  }
}

// out = h2 @ Cd_w^T + Cd_b -> d_out[:,t,:]; also x_{t+1} (bf16). 16x16 tiles (N=144).
__global__ void k_out_gemm(const __bf16* __restrict__ hbf, const __bf16* __restrict__ wCd,
                           const float* __restrict__ Cd_b, float* __restrict__ out,
                           __bf16* __restrict__ xbf, int t) {
  const int lane = threadIdx.x & 31, wave = threadIdx.x >> 5;
  const int half = lane >> 4, l = lane & 15;
  int tile = blockIdx.x * 8 + wave;            // 16 mt * 9 nt = 144 waves
  int nt = tile % (NCD_ / 16), mt = tile / (NCD_ / 16);
  int m0 = mt * 16, n0 = nt * 16;
  const __bf16* Ar = hbf + (size_t)(m0 + l) * H_;
  const __bf16* Wr = wCd + (size_t)(n0 + l) * H_;
  v8f acc = {};
  v16bf af = load_afrag(Ar, 0, half);
  v16bf bf_ = load_bfrag(Wr, 0, half);
  for (int kb = 0; kb < H_; kb += 32) {
    int kn = (kb + 32 < H_) ? kb + 32 : 0;
    v16bf afn = load_afrag(Ar, kn, half);
    v16bf bfn = load_bfrag(Wr, kn, half);
    acc = wmma_bf16(af, bf_, acc);
    af = afn; bf_ = bfn;
  }
  int n = n0 + l;
  if (n < V_) {
    float bias = Cd_b[n];
#pragma unroll
    for (int r = 0; r < 8; ++r) {
      int b = m0 + r + 8 * half;
      float v = acc[r] + bias;
      out[((size_t)b * T_ + t) * V_ + n] = v;
      xbf[(size_t)b * KX_ + n] = f2bf(v);      // pad region [140,160) stays zero
    }
  }
}

// ---------------- attention VALU kernels ----------------

__global__ void k_scores(const float* __restrict__ q, const __bf16* __restrict__ attn,
                         float* __restrict__ scores) {
  __shared__ float qs[A_];
  int b = blockIdx.x, tid = threadIdx.x;
  qs[tid] = q[(size_t)b * A_ + tid];
  __syncthreads();
  for (int p = tid; p < P_; p += 256) {
    const unsigned* row = (const unsigned*)(attn + ((size_t)b * P_ + p) * A_);
    if (p + 256 < P_)
      __builtin_prefetch(row + 256 * (A_ / 2), 0, 1);   // global_prefetch_b8
    float acc = 0.f;
#pragma unroll 8
    for (int a2 = 0; a2 < A_ / 2; ++a2) {
      unsigned w = row[a2];
      acc += qs[2 * a2]     * bflo(w);
      acc += qs[2 * a2 + 1] * bfhi(w);
    }
    scores[(size_t)b * P_ + p] = acc * 0.0625f;         // 1/sqrt(256)
  }
}

__global__ void k_softmax(float* __restrict__ scores) {
  __shared__ float red[8];
  int b = blockIdx.x, tid = threadIdx.x;
  int lane = tid & 31, wave = tid >> 5;
  float* s = scores + (size_t)b * P_;
  float m = -1e30f;
  for (int p = tid; p < P_; p += 256) m = fmaxf(m, s[p]);
#pragma unroll
  for (int o = 16; o > 0; o >>= 1) m = fmaxf(m, __shfl_xor(m, o, 32));
  if (lane == 0) red[wave] = m;
  __syncthreads();
  float mm = red[0];
#pragma unroll
  for (int i = 1; i < 8; ++i) mm = fmaxf(mm, red[i]);
  __syncthreads();
  float sum = 0.f;
  for (int p = tid; p < P_; p += 256) { float e = __expf(s[p] - mm); s[p] = e; sum += e; }
#pragma unroll
  for (int o = 16; o > 0; o >>= 1) sum += __shfl_xor(sum, o, 32);
  if (lane == 0) red[wave] = sum;
  __syncthreads();
  float tot = 0.f;
#pragma unroll
  for (int i = 0; i < 8; ++i) tot += red[i];
  float inv = 1.0f / tot;
  for (int p = tid; p < P_; p += 256) s[p] *= inv;
}

__global__ void k_ctx(const float* __restrict__ w, const __bf16* __restrict__ attn,
                      float* __restrict__ ctx, __bf16* __restrict__ ctxbf) {
  __shared__ float wsm[P_];
  int b = blockIdx.x, tid = threadIdx.x;
  for (int p = tid; p < P_; p += 256) wsm[p] = w[(size_t)b * P_ + p];
  __syncthreads();
  const __bf16* base = attn + (size_t)b * P_ * A_ + tid;
  float acc = 0.f;
#pragma unroll 4
  for (int p = 0; p < P_; ++p) acc += wsm[p] * bf2f(base[(size_t)p * A_]);
  ctx[(size_t)b * A_ + tid] = acc;
  ctxbf[(size_t)b * A_ + tid] = f2bf(acc);
}

// ---------------- LSTM elementwise ----------------

__global__ void k_lstm(const float* __restrict__ gates, float* __restrict__ c,
                       float* __restrict__ h, __bf16* __restrict__ hbf) {
  int idx = blockIdx.x * blockDim.x + threadIdx.x;   // B*H
  int b = idx >> 8, hh = idx & 255;
  const float* g = gates + (size_t)b * G_;
  float gi = g[hh], gf = g[H_ + hh], gg = g[2 * H_ + hh], go = g[3 * H_ + hh];
  float si = 1.f / (1.f + __expf(-gi));
  float sf = 1.f / (1.f + __expf(-gf));
  float so = 1.f / (1.f + __expf(-go));
  float c2 = sf * c[idx] + si * tanhf(gg);
  float h2 = so * tanhf(c2);
  c[idx] = c2;
  h[idx] = h2;
  hbf[idx] = f2bf(h2);
}

// ---------------- launcher ----------------

extern "C" void kernel_launch(void* const* d_in, const int* in_sizes, int n_in,
                              void* d_out, int out_size, void* d_ws, size_t ws_size,
                              hipStream_t stream) {
  const float* img     = (const float*)d_in[0];
  const float* targets = (const float*)d_in[1];
  const float* Ic_w    = (const float*)d_in[2];
  const float* Ic_b    = (const float*)d_in[3];
  const float* Hc_w    = (const float*)d_in[4];
  const float* Hc_b    = (const float*)d_in[5];
  const float* W_ih    = (const float*)d_in[6];
  const float* W_hh    = (const float*)d_in[7];
  const float* b_ih    = (const float*)d_in[8];
  const float* b_hh    = (const float*)d_in[9];
  const float* Cd_w    = (const float*)d_in[10];
  const float* Cd_b    = (const float*)d_in[11];
  float* out = (float*)d_out;

  char* ws = (char*)d_ws;
  size_t off = 0;
  auto alloc = [&](size_t bytes) -> void* {
    void* p = ws + off;
    off = (off + bytes + 255) & ~(size_t)255;
    return p;
  };
  __bf16* attn  = (__bf16*)alloc((size_t)B_ * P_ * A_ * 2);   // 81.3 MB, L2-resident
  __bf16* wIc   = (__bf16*)alloc((size_t)A_ * C_ * 2);
  __bf16* wHc   = (__bf16*)alloc((size_t)A_ * H_ * 2);
  __bf16* wWx   = (__bf16*)alloc((size_t)G_ * KX_ * 2);
  __bf16* wWc   = (__bf16*)alloc((size_t)G_ * A_ * 2);
  __bf16* wWhh  = (__bf16*)alloc((size_t)G_ * H_ * 2);
  __bf16* wCd   = (__bf16*)alloc((size_t)NCD_ * H_ * 2);
  float*  h     = (float*)alloc((size_t)B_ * H_ * 4);
  float*  c     = (float*)alloc((size_t)B_ * H_ * 4);
  __bf16* hbf   = (__bf16*)alloc((size_t)B_ * H_ * 2);
  __bf16* xbf   = (__bf16*)alloc((size_t)B_ * KX_ * 2);
  float*  q     = (float*)alloc((size_t)B_ * A_ * 4);
  float*  sc    = (float*)alloc((size_t)B_ * P_ * 4);
  float*  ctx   = (float*)alloc((size_t)B_ * A_ * 4);
  __bf16* ctxbf = (__bf16*)alloc((size_t)B_ * A_ * 2);
  float*  gates = (float*)alloc((size_t)B_ * G_ * 4);

  k_cvt<<<(A_ * C_ + 255) / 256, 256, 0, stream>>>(Ic_w, wIc, A_, C_, C_, 0, A_, C_);
  k_cvt<<<(A_ * H_ + 255) / 256, 256, 0, stream>>>(Hc_w, wHc, A_, H_, H_, 0, A_, H_);
  k_cvt<<<(G_ * KX_ + 255) / 256, 256, 0, stream>>>(W_ih, wWx, G_, V_, V_ + A_, 0, G_, KX_);
  k_cvt<<<(G_ * A_ + 255) / 256, 256, 0, stream>>>(W_ih, wWc, G_, A_, V_ + A_, V_, G_, A_);
  k_cvt<<<(G_ * H_ + 255) / 256, 256, 0, stream>>>(W_hh, wWhh, G_, H_, H_, 0, G_, H_);
  k_cvt<<<(NCD_ * H_ + 255) / 256, 256, 0, stream>>>(Cd_w, wCd, V_, H_, H_, 0, NCD_, H_);
  k_init_state<<<(B_ * H_ + 255) / 256, 256, 0, stream>>>(targets, h, c, hbf, xbf);

  // Big WMMA GEMM: (B*P/16) M-tiles x (A/64) N-groups, 8 waves/block.
  k_attn_gemm<<<(B_ * P_ / 16) * (A_ / 64) / 8, 256, 0, stream>>>(img, wIc, Ic_b, attn);

  for (int t = 0; t < T_; ++t) {
    k_q_gemm<<<(B_ / 16) * (A_ / 64) / 8, 256, 0, stream>>>(hbf, wHc, Hc_b, q);
    k_scores<<<B_, 256, 0, stream>>>(q, attn, sc);
    k_softmax<<<B_, 256, 0, stream>>>(sc);
    k_ctx<<<B_, 256, 0, stream>>>(sc, attn, ctx, ctxbf);
    k_gates_gemm<<<(B_ / 16) * (G_ / 64) / 8, 256, 0, stream>>>(xbf, ctxbf, hbf, wWx, wWc,
                                                                wWhh, b_ih, b_hh, gates);
    k_lstm<<<(B_ * H_) / 256, 256, 0, stream>>>(gates, c, h, hbf);
    k_out_gemm<<<(B_ / 16) * (NCD_ / 16) / 8, 256, 0, stream>>>(hbf, wCd, Cd_b, out, xbf, t);
  }
}
